// LSTM2d_77464030151092
// MI455X (gfx1250) — compile-verified, matrix operand order checked by reading, not run on hardware
//
#include <hip/hip_runtime.h>
#include <hip/hip_bf16.h>

// Problem constants (from reference)
#define L_SEQ 48
#define BATCH 32
#define D_EMB 512
#define E_DIM 512
#define S_DIM 512
#define VOCAB 32000
#define ROWS  (L_SEQ * BATCH)        // 1536
#define XK    (2 * E_DIM + D_EMB)    // 1536  (x_cell width)
#define G4    (4 * S_DIM)            // 2048  (packed gates i,f,o,g)

typedef __bf16 bf16_t;
typedef __attribute__((ext_vector_type(16))) __bf16 bf16x16;
typedef __attribute__((ext_vector_type(8)))  __bf16 bf16x8;
typedef __attribute__((ext_vector_type(8)))  float  f32x8;

__device__ __forceinline__ bf16_t f2bf(float f) {
  unsigned u = __builtin_bit_cast(unsigned, f);
  unsigned r = u + 0x7FFFu + ((u >> 16) & 1u);   // round-to-nearest-even
  unsigned short s = (unsigned short)(r >> 16);
  return __builtin_bit_cast(bf16_t, s);
}
__device__ __forceinline__ float sigm(float x) { return 1.0f / (1.0f + __expf(-x)); }

// Fragment load per CDNA5 ISA 7.12.2 (16-bit A/B 16x32): lane = fr + 16*h holds
// row fr; K-slots {k..k+7} at koff=8*h and {k+16..k+23} at koff+16 -> two b128s.
__device__ __forceinline__ bf16x16 ldfrag(const bf16_t* __restrict__ p, int k) {
  bf16x8 lo = *(const bf16x8*)(p + k);
  bf16x8 hi = *(const bf16x8*)(p + k + 16);
  return __builtin_shufflevector(lo, hi, 0,1,2,3,4,5,6,7,8,9,10,11,12,13,14,15);
}

// D layout: VGPR r, lanes 0-15 -> row m+r, lanes 16-31 -> row m+r+8; col = n+(lane&15)
__device__ __forceinline__ void store_tile(float* __restrict__ C, int ldc,
                                           const float* __restrict__ Cadd,
                                           const float* __restrict__ bias,
                                           int m, int n, int lane, f32x8 acc) {
  const int col   = n + (lane & 15);
  const int rbase = m + ((lane >> 4) << 3);
  const float badd = bias ? bias[col] : 0.0f;
#pragma unroll
  for (int r = 0; r < 8; ++r) {
    size_t idx = (size_t)(rbase + r) * ldc + col;
    float v = acc[r] + badd;
    if (Cadd) v += Cadd[idx];
    C[idx] = v;
  }
}

// ---------------------------------------------------------------------------
// bf16 WMMA GEMM, 2x2 register-tiled:
//   C[M,N](f32) = A[M,K](bf16,row-major) @ W[N,K].T (+ bias[col]) (+ Cadd)
// Block: 128 threads = 4 waves. Each wave computes a 32x32 C tile (4 WMMAs per
// 32-wide k-step from 2 A-frags + 2 B-frags -> 2 b128 loads per WMMA).
// Block tile = 32 rows x 128 cols.  grid = (N/128, M/32).
// Requires: M % 32 == 0, N % 128 == 0, K % 32 == 0 (all true here).
// ---------------------------------------------------------------------------
__global__ __launch_bounds__(128)
void gemm_bf16_wmma(const bf16_t* __restrict__ A, int lda,
                    const bf16_t* __restrict__ W, int ldw,
                    float* __restrict__ C, int ldc,
                    const float* __restrict__ Cadd,
                    const float* __restrict__ bias,
                    int K)
{
  const int lane = threadIdx.x & 31;
  const int wave = threadIdx.x >> 5;
  const int m0   = blockIdx.y << 5;                    // 32 rows per block
  const int n0   = (blockIdx.x << 7) + (wave << 5);    // 32 cols per wave

  const int fr   = lane & 15;
  const int koff = (lane >> 4) << 3;                   // 0 or 8

  const bf16_t* pA0 = A + (size_t)(m0 + fr) * lda + koff;
  const bf16_t* pA1 = pA0 + (size_t)16 * lda;
  const bf16_t* pW0 = W + (size_t)(n0 + fr) * ldw + koff;
  const bf16_t* pW1 = pW0 + (size_t)16 * ldw;

  f32x8 acc00 = {0.f,0.f,0.f,0.f,0.f,0.f,0.f,0.f};
  f32x8 acc01 = acc00, acc10 = acc00, acc11 = acc00;

  for (int k = 0; k < K; k += 32) {
    bf16x16 a0 = ldfrag(pA0, k);
    bf16x16 a1 = ldfrag(pA1, k);
    bf16x16 b0 = ldfrag(pW0, k);
    bf16x16 b1 = ldfrag(pW1, k);
    acc00 = __builtin_amdgcn_wmma_f32_16x16x32_bf16(false, a0, false, b0, (short)0, acc00, false, false);
    acc01 = __builtin_amdgcn_wmma_f32_16x16x32_bf16(false, a0, false, b1, (short)0, acc01, false, false);
    acc10 = __builtin_amdgcn_wmma_f32_16x16x32_bf16(false, a1, false, b0, (short)0, acc10, false, false);
    acc11 = __builtin_amdgcn_wmma_f32_16x16x32_bf16(false, a1, false, b1, (short)0, acc11, false, false);
  }

  store_tile(C, ldc, Cadd, bias, m0,      n0,      lane, acc00);
  store_tile(C, ldc, Cadd, bias, m0,      n0 + 16, lane, acc01);
  store_tile(C, ldc, Cadd, bias, m0 + 16, n0,      lane, acc10);
  store_tile(C, ldc, Cadd, bias, m0 + 16, n0 + 16, lane, acc11);
}

// ---------------------------------------------------------------------------
// Utility kernels
// ---------------------------------------------------------------------------
__global__ void k_zero_u32(unsigned* __restrict__ p, long n) {
  long i = (long)blockIdx.x * blockDim.x + threadIdx.x;
  if (i < n) p[i] = 0u;
}
__global__ void k_f2bf(const float* __restrict__ in, bf16_t* __restrict__ out, long n) {
  long i = (long)blockIdx.x * blockDim.x + threadIdx.x;
  if (i < n) out[i] = f2bf(in[i]);
}
// encoder input embedding gather -> bf16 [1536, 512]
__global__ void k_gather_xe(const int* __restrict__ x, const float* __restrict__ emb,
                            bf16_t* __restrict__ xe) {
  long i = (long)blockIdx.x * blockDim.x + threadIdx.x;
  if (i >= (long)ROWS * D_EMB) return;
  int row = (int)(i >> 9), e = (int)(i & 511);
  xe[i] = f2bf(emb[(size_t)x[row] * D_EMB + e]);
}
// teacher-forced output embedding -> Xcell columns [1024,1536)
__global__ void k_gather_yemb(const int* __restrict__ y, const float* __restrict__ emb,
                              bf16_t* __restrict__ Xcell) {
  long i = (long)blockIdx.x * blockDim.x + threadIdx.x;
  if (i >= (long)ROWS * D_EMB) return;
  int row = (int)(i >> 9), e = (int)(i & 511);
  int l = row >> 5, b = row & 31;
  int tok = (l == 0) ? 0 : y[(l - 1) * BATCH + b];   // START_TOKEN = 0
  Xcell[(size_t)row * XK + 2 * E_DIM + e] = f2bf(emb[(size_t)tok * D_EMB + e]);
}
// pack cell_Wx rows {i,f,o,g} (drop dead lambda gate), fp32 -> bf16
__global__ void k_pack_cWx(const float* __restrict__ Wx, bf16_t* __restrict__ out) {
  long i = (long)blockIdx.x * blockDim.x + threadIdx.x;
  if (i >= (long)G4 * XK) return;
  int j = (int)(i / XK), k = (int)(i % XK);
  int src = (j < 3 * S_DIM) ? j : j + S_DIM;        // skip lambda rows [1536,2048)
  out[i] = f2bf(Wx[(size_t)src * XK + k]);
}
// fold cell_Ws:  s_hor==s_ver  ->  Wf[j,k] = Ws[j,k] + Ws[j,k+512]; pack i,f,o,g
__global__ void k_pack_cWs(const float* __restrict__ Ws, bf16_t* __restrict__ out) {
  long i = (long)blockIdx.x * blockDim.x + threadIdx.x;
  if (i >= (long)G4 * S_DIM) return;
  int j = (int)(i >> 9), k = (int)(i & 511);
  int src = (j < 3 * S_DIM) ? j : j + S_DIM;
  const float* r = Ws + (size_t)src * (2 * S_DIM);
  out[i] = f2bf(r[k] + r[S_DIM + k]);
}
__global__ void k_pack_cb(const float* __restrict__ b, float* __restrict__ out) {
  int j = blockIdx.x * blockDim.x + threadIdx.x;
  if (j >= G4) return;
  int src = (j < 3 * S_DIM) ? j : j + S_DIM;
  out[j] = b[src];
}

// Encoder activation, fwd+bwd in one launch. torch gate order: i,f,g,o.
__global__ void k_enc_act(const float* __restrict__ Pf, const float* __restrict__ Pb,
                          float* __restrict__ cf, float* __restrict__ cb,
                          bf16_t* __restrict__ hf, bf16_t* __restrict__ hb,
                          bf16_t* __restrict__ Xcell, int t)
{
  int idx = blockIdx.x * blockDim.x + threadIdx.x;       // 2*32*512
  if (idx >= 2 * BATCH * E_DIM) return;
  int dir = idx >= BATCH * E_DIM;
  int j = idx - dir * BATCH * E_DIM;
  int b = j >> 9, e = j & 511;
  const float* p = (dir ? Pb : Pf) + (size_t)b * G4;
  float* c = dir ? cb : cf;
  float pi = p[e], pf_ = p[512 + e], pg = p[1024 + e], po = p[1536 + e];
  float cn = sigm(pf_) * c[j] + sigm(pi) * tanhf(pg);
  c[j] = cn;
  float h = sigm(po) * tanhf(cn);
  bf16_t hb16 = f2bf(h);
  (dir ? hb : hf)[j] = hb16;
  int l = dir ? (L_SEQ - 1 - t) : t;
  Xcell[(size_t)(l * BATCH + b) * XK + dir * E_DIM + e] = hb16;
}

// 2D-cell activation.  packed gate order: i,f,o,g.  c_hor==c_ver => c=f*c+i*tanh(g)
__global__ void k_dec_act(const float* __restrict__ P,
                          float* __restrict__ c, bf16_t* __restrict__ s)
{
  int idx = blockIdx.x * blockDim.x + threadIdx.x;       // 1536*512
  if (idx >= ROWS * S_DIM) return;
  int r = idx >> 9, e = idx & 511;
  const float* p = P + (size_t)r * G4;
  float pi = p[e], pf_ = p[512 + e], po = p[1024 + e], pg = p[1536 + e];
  float cn = sigm(pf_) * c[idx] + sigm(pi) * tanhf(pg);
  c[idx] = cn;
  s[idx] = f2bf(sigm(po) * tanhf(cn));
}

// In-place row softmax over VOCAB, one block per row
__global__ __launch_bounds__(256)
void k_softmax(float* __restrict__ out)
{
  __shared__ float red[256];
  float* p = out + (size_t)blockIdx.x * VOCAB;
  int tid = threadIdx.x;
  float m = -3.4e38f;
  for (int i = tid; i < VOCAB; i += 256) m = fmaxf(m, p[i]);
  red[tid] = m; __syncthreads();
  for (int s = 128; s > 0; s >>= 1) {
    if (tid < s) red[tid] = fmaxf(red[tid], red[tid + s]);
    __syncthreads();
  }
  m = red[0]; __syncthreads();
  float sum = 0.f;
  for (int i = tid; i < VOCAB; i += 256) { float e = __expf(p[i] - m); p[i] = e; sum += e; }
  red[tid] = sum; __syncthreads();
  for (int s = 128; s > 0; s >>= 1) {
    if (tid < s) red[tid] += red[tid + s];
    __syncthreads();
  }
  float inv = 1.f / red[0];
  for (int i = tid; i < VOCAB; i += 256) p[i] *= inv;
}

// ---------------------------------------------------------------------------
extern "C" void kernel_launch(void* const* d_in, const int* in_sizes, int n_in,
                              void* d_out, int out_size, void* d_ws, size_t ws_size,
                              hipStream_t stream)
{
  (void)in_sizes; (void)n_in; (void)out_size; (void)ws_size;
  const int*   x        = (const int*)  d_in[0];
  const int*   y        = (const int*)  d_in[1];
  const float* emb_in   = (const float*)d_in[2];
  const float* emb_out  = (const float*)d_in[3];
  const float* eWih_f   = (const float*)d_in[4];
  const float* eWhh_f   = (const float*)d_in[5];
  const float* eb_f     = (const float*)d_in[6];
  const float* eWih_b   = (const float*)d_in[7];
  const float* eWhh_b   = (const float*)d_in[8];
  const float* eb_b     = (const float*)d_in[9];
  const float* cWx      = (const float*)d_in[10];
  const float* cWs      = (const float*)d_in[11];
  const float* cb       = (const float*)d_in[12];
  const float* outW     = (const float*)d_in[13];
  const float* outb     = (const float*)d_in[14];
  float* out = (float*)d_out;

  // workspace carve-up (256B aligned)
  char* ws = (char*)d_ws; size_t off = 0;
  auto carve = [&](size_t bytes) {
    void* p = ws + off; off += (bytes + 255) & ~(size_t)255; return p;
  };
  bf16_t* xe     = (bf16_t*)carve((size_t)ROWS * D_EMB * 2);
  bf16_t* wih_f  = (bf16_t*)carve((size_t)G4 * D_EMB * 2);
  bf16_t* whh_f  = (bf16_t*)carve((size_t)G4 * E_DIM * 2);
  bf16_t* wih_b  = (bf16_t*)carve((size_t)G4 * D_EMB * 2);
  bf16_t* whh_b  = (bf16_t*)carve((size_t)G4 * E_DIM * 2);
  bf16_t* outw   = (bf16_t*)carve((size_t)VOCAB * S_DIM * 2);
  bf16_t* cwx_p  = (bf16_t*)carve((size_t)G4 * XK * 2);
  bf16_t* cws_p  = (bf16_t*)carve((size_t)G4 * S_DIM * 2);
  float*  cb_p   = (float*) carve((size_t)G4 * 4);
  float*  Xf     = (float*) carve((size_t)ROWS * G4 * 4);
  float*  Xb     = (float*) carve((size_t)ROWS * G4 * 4);
  bf16_t* Xcell  = (bf16_t*)carve((size_t)ROWS * XK * 2);
  float*  Apre   = (float*) carve((size_t)ROWS * G4 * 4);
  float*  Pbuf   = (float*) carve((size_t)ROWS * G4 * 4);
  float*  Pf     = (float*) carve((size_t)BATCH * G4 * 4);
  float*  Pb     = (float*) carve((size_t)BATCH * G4 * 4);
  float*  cf     = (float*) carve((size_t)BATCH * E_DIM * 4);
  float*  cbk    = (float*) carve((size_t)BATCH * E_DIM * 4);
  bf16_t* hf     = (bf16_t*)carve((size_t)BATCH * E_DIM * 2);
  bf16_t* hb     = (bf16_t*)carve((size_t)BATCH * E_DIM * 2);
  float*  cdec   = (float*) carve((size_t)ROWS * S_DIM * 4);
  bf16_t* sbf    = (bf16_t*)carve((size_t)ROWS * S_DIM * 2);

  auto blocks = [](long n) { return dim3((unsigned)((n + 255) / 256)); };

  // --- deterministic state init (graph replays: must self-zero) ---
  k_zero_u32<<<blocks(BATCH * E_DIM), 256, 0, stream>>>((unsigned*)cf,  BATCH * E_DIM);
  k_zero_u32<<<blocks(BATCH * E_DIM), 256, 0, stream>>>((unsigned*)cbk, BATCH * E_DIM);
  k_zero_u32<<<blocks(BATCH * E_DIM / 2), 256, 0, stream>>>((unsigned*)hf, BATCH * E_DIM / 2);
  k_zero_u32<<<blocks(BATCH * E_DIM / 2), 256, 0, stream>>>((unsigned*)hb, BATCH * E_DIM / 2);
  k_zero_u32<<<blocks((long)ROWS * S_DIM), 256, 0, stream>>>((unsigned*)cdec, (long)ROWS * S_DIM);
  k_zero_u32<<<blocks((long)ROWS * S_DIM / 2), 256, 0, stream>>>((unsigned*)sbf, (long)ROWS * S_DIM / 2);

  // --- gathers + weight conversion / packing ---
  k_gather_xe  <<<blocks((long)ROWS * D_EMB), 256, 0, stream>>>(x, emb_in, xe);
  k_gather_yemb<<<blocks((long)ROWS * D_EMB), 256, 0, stream>>>(y, emb_out, Xcell);
  k_f2bf<<<blocks((long)G4 * D_EMB), 256, 0, stream>>>(eWih_f, wih_f, (long)G4 * D_EMB);
  k_f2bf<<<blocks((long)G4 * E_DIM), 256, 0, stream>>>(eWhh_f, whh_f, (long)G4 * E_DIM);
  k_f2bf<<<blocks((long)G4 * D_EMB), 256, 0, stream>>>(eWih_b, wih_b, (long)G4 * D_EMB);
  k_f2bf<<<blocks((long)G4 * E_DIM), 256, 0, stream>>>(eWhh_b, whh_b, (long)G4 * E_DIM);
  k_f2bf<<<blocks((long)VOCAB * S_DIM), 256, 0, stream>>>(outW, outw, (long)VOCAB * S_DIM);
  k_pack_cWx<<<blocks((long)G4 * XK), 256, 0, stream>>>(cWx, cwx_p);
  k_pack_cWs<<<blocks((long)G4 * S_DIM), 256, 0, stream>>>(cWs, cws_p);
  k_pack_cb <<<blocks(G4), 256, 0, stream>>>(cb, cb_p);

  // --- encoder: time-invariant input projections (one big GEMM each) ---
  dim3 gBig(G4 / 128, ROWS / 32);          // (16, 48)
  gemm_bf16_wmma<<<gBig, 128, 0, stream>>>(xe, D_EMB, wih_f, D_EMB, Xf, G4, nullptr, eb_f, D_EMB);
  gemm_bf16_wmma<<<gBig, 128, 0, stream>>>(xe, D_EMB, wih_b, D_EMB, Xb, G4, nullptr, eb_b, D_EMB);

  // --- encoder recurrence: 48 steps, fwd+bwd small GEMMs + fused activation ---
  dim3 gEnc(G4 / 128, BATCH / 32);         // (16, 1)
  for (int t = 0; t < L_SEQ; ++t) {
    gemm_bf16_wmma<<<gEnc, 128, 0, stream>>>(hf, E_DIM, whh_f, E_DIM, Pf, G4,
                                             Xf + (size_t)t * BATCH * G4, nullptr, E_DIM);
    gemm_bf16_wmma<<<gEnc, 128, 0, stream>>>(hb, E_DIM, whh_b, E_DIM, Pb, G4,
                                             Xb + (size_t)(L_SEQ - 1 - t) * BATCH * G4, nullptr, E_DIM);
    k_enc_act<<<blocks(2 * BATCH * E_DIM), 256, 0, stream>>>(Pf, Pb, cf, cbk, hf, hb, Xcell, t);
  }

  // --- 2D cell: iteration-invariant term Apre = Xcell @ cWx_packed.T + b ---
  gemm_bf16_wmma<<<gBig, 128, 0, stream>>>(Xcell, XK, cwx_p, XK, Apre, G4, nullptr, cb_p, XK);

  // --- 48 recurrent steps: P = s @ cWs_folded.T + Apre; gate math ---
  for (int t = 0; t < L_SEQ; ++t) {
    gemm_bf16_wmma<<<gBig, 128, 0, stream>>>(sbf, S_DIM, cws_p, S_DIM, Pbuf, G4,
                                             Apre, nullptr, S_DIM);
    k_dec_act<<<blocks((long)ROWS * S_DIM), 256, 0, stream>>>(Pbuf, cdec, sbf);
  }

  // --- output projection straight into d_out, then in-place softmax ---
  dim3 gOut(VOCAB / 128, ROWS / 32);       // (250, 48)
  gemm_bf16_wmma<<<gOut, 128, 0, stream>>>(sbf, S_DIM, outw, S_DIM, out, VOCAB,
                                           nullptr, outb, S_DIM);
  k_softmax<<<ROWS, 256, 0, stream>>>(out);
}
